// TransformerLayer_19610820674179
// MI455X (gfx1250) — matrix operand order repaired and checked by reading, not compiled, not run on hardware
//
#include <hip/hip_runtime.h>

typedef __bf16 bf16_t;
typedef __attribute__((ext_vector_type(16))) __bf16 v16bf;
typedef __attribute__((ext_vector_type(8)))  __bf16 v8bf;
typedef __attribute__((ext_vector_type(8)))  float  v8f;
typedef __attribute__((ext_vector_type(4)))  unsigned int v4u;
typedef __attribute__((ext_vector_type(4)))  int v4i;
typedef __attribute__((ext_vector_type(8)))  int v8i;

union ABFrag { v16bf v; v8bf h[2]; };

// ---------------------------------------------------------------------------
// CDNA5 WMMA fragment loaders (wave32, 16x16x32 bf16).
// A (16x32, MxK): lane r=lane&15 holds row r; lanes 0-15 hold K 0..15,
// lanes 16-31 hold K 16..31  -> chunks at kh=(lane>>4)*16 and kh+8.
// B (32x16, KxN): lane c=lane&15 holds column c; lanes 0-15 hold K 0..15,
// lanes 16-31 hold K 16..31 -> 16 contiguous K values at kh=(lane>>4)*16,
// sourced from a B^T (N-major, K-contiguous) tile.
// C/D (16x16 f32): lane n=lane&15 is column; VGPR r is row r (lanes 0-15)
// or row 8+r (lanes 16-31).
// ---------------------------------------------------------------------------
__device__ __forceinline__ v16bf load_a_frag(const bf16_t* base, int ld) {
  const int lane = threadIdx.x & 31;
  const int r = lane & 15;
  const int kh = (lane >> 4) * 16;
  ABFrag f;
  f.h[0] = *(const v8bf*)(base + r * ld + kh);
  f.h[1] = *(const v8bf*)(base + r * ld + kh + 8);
  return f.v;
}

__device__ __forceinline__ v16bf load_b_frag(const bf16_t* base, int ld) {
  const int lane = threadIdx.x & 31;
  const int c = lane & 15;
  const int kh = (lane >> 4) * 16;
  return *(const v16bf*)(base + c * ld + kh);
}

__device__ __forceinline__ v8f wmma_bf16(v16bf a, v16bf b, v8f c) {
  return __builtin_amdgcn_wmma_f32_16x16x32_bf16(false, a, false, b, (short)0, c,
                                                 false, false);
}

// Async global -> LDS copy, 16B per lane, tracked by ASYNCcnt (CDNA5 §10.7).
// GVS addressing: mem = saddr64 + vaddr32; LDS dest = per-lane u32 offset.
__device__ __forceinline__ void async_b128(unsigned int lds_addr,
                                           unsigned int voff_bytes,
                                           unsigned long long base) {
  asm volatile("global_load_async_to_lds_b128 %0, %1, %2"
               :: "v"(lds_addr), "v"(voff_bytes), "s"(base)
               : "memory");
}

// Tensor Data Mover: DMA a 2D tile (tile_d0 x tile_d1 elements of 2 bytes,
// row stride stride0 elements) from global to LDS. D# per CDNA5 ISA ch.8:
//   group0: count=1 | lds_addr | global_addr(57b) | type=2
//   group1: data_size=1(2B); tensor_dim0/1; tile_dim0/1; tensor_dim0_stride
//   groups 2/3: zero (dims 3..5 unused)
// Tracked by TENSORcnt; issue from one wave only.
__device__ __forceinline__ void tdm_load_2d(unsigned int lds_addr,
                                            unsigned long long gaddr,
                                            unsigned int tensor_d0,
                                            unsigned int tensor_d1,
                                            unsigned int tile_d0,
                                            unsigned int tile_d1,
                                            unsigned long long stride0) {
  v4u g0;
  g0.x = 1u;                                   // count=1, load, user D#
  g0.y = lds_addr;                             // LDS byte offset
  g0.z = (unsigned int)gaddr;                  // global_addr[31:0]
  g0.w = (unsigned int)((gaddr >> 32) & 0x1FFFFFFu) | 0x80000000u;  // type=2
  v8i g1;
  g1[0] = 0x10000;                             // data_size=1 -> 2 bytes
  g1[1] = (int)((tensor_d0 & 0xFFFFu) << 16);  // tensor_dim0[15:0]
  g1[2] = (int)((tensor_d0 >> 16) | ((tensor_d1 & 0xFFFFu) << 16));
  g1[3] = (int)((tensor_d1 >> 16) | (tile_d0 << 16));
  g1[4] = (int)(tile_d1 & 0xFFFFu);            // tile_dim1, tile_dim2=0
  g1[5] = (int)(unsigned int)(stride0 & 0xFFFFFFFFu);
  g1[6] = (int)(unsigned int)((stride0 >> 32) & 0xFFFFu);
  g1[7] = 0;
  v4i z4 = {0, 0, 0, 0};
#if defined(__clang_major__) && (__clang_major__ >= 23)
  v8i z8 = {0, 0, 0, 0, 0, 0, 0, 0};
  __builtin_amdgcn_tensor_load_to_lds(g0, g1, z4, z4, z8, 0);
#else
  __builtin_amdgcn_tensor_load_to_lds(g0, g1, z4, z4, 0);
#endif
}

// ---------------------------------------------------------------------------
// Elementwise / packing kernels
// ---------------------------------------------------------------------------
__global__ __launch_bounds__(256) void cast_bf16_kernel(const float* __restrict__ in,
                                                        bf16_t* __restrict__ out,
                                                        int n) {
  int i = blockIdx.x * 256 + threadIdx.x;
  if (i < n) out[i] = (bf16_t)in[i];
}

// in [K,N] row-major f32 -> out [N,K] row-major bf16 (B^T layout)
__global__ __launch_bounds__(256) void transpose_cast_kernel(const float* __restrict__ in,
                                                             bf16_t* __restrict__ out,
                                                             int K, int N) {
  int i = blockIdx.x * 256 + threadIdx.x;
  if (i >= K * N) return;
  int k = i / N, n = i - k * N;
  out[(size_t)n * K + k] = (bf16_t)in[i];
}

// w [H,D,dh] f32 -> dstT[(h*dh+e)*D + k] bf16 (rows = output cols of QKV gemm)
__global__ __launch_bounds__(256) void pack_headw_kernel(const float* __restrict__ w,
                                                         bf16_t* __restrict__ dstT,
                                                         int D, int dh, int n) {
  int i = blockIdx.x * 256 + threadIdx.x;
  if (i >= n) return;
  int e = i % dh;
  int k = (i / dh) % D;
  int h = i / (dh * D);
  dstT[(size_t)(h * dh + e) * D + k] = (bf16_t)w[i];
}

// ---------------------------------------------------------------------------
// Generic bf16 GEMM:  C[M,N] = A[M,K] @ B[K,N]  with B given transposed (BT[N,K]).
// Block: 256 threads = 8 waves, tile 128(M) x 64(N) x 32(K). Waves in 4x2 grid,
// each wave computes 32x32 = 4 WMMA per K-step. Double-buffered staging:
//   A tile: GLOBAL_LOAD_ASYNC_TO_LDS_B128 (2 per thread), ASYNCcnt
//   B tile: one TENSOR_LOAD_TO_LDS descriptor issued by wave 0, TENSORcnt
// both overlapped with the WMMA work of the previous stage.
// ---------------------------------------------------------------------------
__global__ __launch_bounds__(256) void gemm_bf16_kernel(
    const bf16_t* __restrict__ A, const bf16_t* __restrict__ BT,
    float* __restrict__ Cf, bf16_t* __restrict__ Cb,
    const float* __restrict__ bias, int M, int N, int K, int relu) {
  constexpr int BM = 128, BN = 64, BK = 32;
  __shared__ __align__(64) bf16_t As[2][BM * BK];
  __shared__ __align__(64) bf16_t Bs[2][BN * BK];
  const int tid = threadIdx.x;
  const int bm = blockIdx.y * BM;
  const int bn = blockIdx.x * BN;
  const int wave = tid >> 5, lane = tid & 31;
  const int wm = (wave >> 1) * 32;  // 0,32,64,96
  const int wn = (wave & 1) * 32;   // 0,32

  // Async staging map for A: 512 16B-chunks, 2 per thread.
  const int ca1 = tid + 256;
  const int ra0 = tid >> 2, ka0 = (tid & 3) * 8;
  const int ra1 = ca1 >> 2, ka1 = (ca1 & 3) * 8;

  const unsigned long long abase = (unsigned long long)(size_t)A;
  const unsigned long long bbase = (unsigned long long)(size_t)BT;
  const unsigned int avo0 = (unsigned int)(((size_t)(bm + ra0) * K + ka0) * 2);
  const unsigned int avo1 = (unsigned int)(((size_t)(bm + ra1) * K + ka1) * 2);
  const unsigned long long bga = bbase + ((size_t)bn * K) * 2;  // tile row 0, k 0
  unsigned int alds0[2], alds1[2], blds[2];
#pragma unroll
  for (int i = 0; i < 2; ++i) {
    alds0[i] = (unsigned int)(size_t)(&As[i][ra0 * BK + ka0]);
    alds1[i] = (unsigned int)(size_t)(&As[i][ra1 * BK + ka1]);
    blds[i] = (unsigned int)(size_t)(&Bs[i][0]);
  }

  v8f acc00 = {}, acc01 = {}, acc10 = {}, acc11 = {};
  const int nk = K / BK;
  const bool issuer = (wave == 0);

  // prologue: stage 0 into buffer 0
  async_b128(alds0[0], avo0, abase);
  async_b128(alds1[0], avo1, abase);
  if (issuer)
    tdm_load_2d(blds[0], bga, (unsigned int)K, (unsigned int)N, BK, BN,
                (unsigned long long)K);

  for (int k = 0; k < nk; ++k) {
    const int cur = k & 1;
    if (k + 1 < nk) {
      const unsigned int koff = (unsigned int)((k + 1) * BK * 2);
      const int nxt = cur ^ 1;
      async_b128(alds0[nxt], avo0 + koff, abase);
      async_b128(alds1[nxt], avo1 + koff, abase);
      if (issuer)
        tdm_load_2d(blds[nxt], bga + koff, (unsigned int)K, (unsigned int)N,
                    BK, BN, (unsigned long long)K);
      // stage k complete when <= (stage k+1 in-flight) remain; both counters
      // complete in issue order within a wave.
      asm volatile("s_wait_asynccnt 0x2" ::: "memory");
      if (issuer) __builtin_amdgcn_s_wait_tensorcnt((short)1);
    } else {
      asm volatile("s_wait_asynccnt 0x0" ::: "memory");
      if (issuer) __builtin_amdgcn_s_wait_tensorcnt((short)0);
    }
    __syncthreads();  // all waves' stage-k data resident in LDS
    const bf16_t* ab = &As[cur][0];
    const bf16_t* bb = &Bs[cur][0];
    v16bf a0 = load_a_frag(ab + wm * BK, BK);
    v16bf a1 = load_a_frag(ab + (wm + 16) * BK, BK);
    v16bf b0 = load_b_frag(bb + wn * BK, BK);
    v16bf b1 = load_b_frag(bb + (wn + 16) * BK, BK);
    acc00 = wmma_bf16(a0, b0, acc00);
    acc01 = wmma_bf16(a0, b1, acc01);
    acc10 = wmma_bf16(a1, b0, acc10);
    acc11 = wmma_bf16(a1, b1, acc11);
    __syncthreads();  // done reading buf[cur] before stage k+2 overwrites it
  }

  const int cn = lane & 15;
  const int rbase = (lane >> 4) * 8;
#pragma unroll
  for (int r = 0; r < 8; ++r) {
#pragma unroll
    for (int mi = 0; mi < 2; ++mi) {
      const int row = bm + wm + mi * 16 + rbase + r;
#pragma unroll
      for (int ni = 0; ni < 2; ++ni) {
        const int col = bn + wn + ni * 16 + cn;
        float v = (mi == 0) ? (ni == 0 ? acc00[r] : acc01[r])
                            : (ni == 0 ? acc10[r] : acc11[r]);
        if (bias) v += bias[col];
        if (relu) v = fmaxf(v, 0.f);
        if (Cf) Cf[(size_t)row * N + col] = v;
        if (Cb) Cb[(size_t)row * N + col] = (bf16_t)v;
      }
    }
  }
}

// ---------------------------------------------------------------------------
// Causal flash attention. qkv: [B*S, 3072] bf16 with Q at col h*64,
// K at 1024+h*64, V at 2048+h*64. One wave handles one 16-row query block,
// iterating 32-key blocks with online softmax. out: [B*S, 1024] bf16 (heads
// concatenated).
// ---------------------------------------------------------------------------
__global__ __launch_bounds__(256) void attn_kernel(const bf16_t* __restrict__ qkv,
                                                   bf16_t* __restrict__ out) {
  constexpr int S = 2048, NQ = 3072, DH = 64, DM = 1024;
  __shared__ bf16_t Ps[8][16 * 32];
  const int wave = threadIdx.x >> 5;
  const int lane = threadIdx.x & 31;
  const int wid = blockIdx.x * 8 + wave;
  const int qb = wid & 127;          // S/16 = 128 query blocks
  const int h = (wid >> 7) & 15;
  const int b = wid >> 11;
  const int lr = lane & 15;
  const int lh = lane >> 4;

  const bf16_t* qbase = qkv + (size_t)b * S * NQ + h * DH;
  const bf16_t* kbase = qbase + 1024;
  const bf16_t* vbase = qbase + 2048;

  // Q fragments for K-dim 0..31 and 32..63
  ABFrag qa0, qa1;
  {
    const bf16_t* qrow = qbase + (size_t)(qb * 16 + lr) * NQ;
    const int kh = lh * 16;
    qa0.h[0] = *(const v8bf*)(qrow + kh);
    qa0.h[1] = *(const v8bf*)(qrow + kh + 8);
    qa1.h[0] = *(const v8bf*)(qrow + 32 + kh);
    qa1.h[1] = *(const v8bf*)(qrow + 32 + kh + 8);
  }

  float m[8], l[8];
#pragma unroll
  for (int r = 0; r < 8; ++r) { m[r] = -1e30f; l[r] = 0.f; }
  v8f o0 = {}, o1 = {}, o2 = {}, o3 = {};

  const int jend = (qb * 16 + 15) >> 5;
  const float scale = 0.125f;  // 1/sqrt(64)
  bf16_t* ps = &Ps[wave][0];

  for (int j = 0; j <= jend; ++j) {
    const int kb = j * 32;
    v8f s0 = {}, s1 = {};
#pragma unroll
    for (int kk = 0; kk < 2; ++kk) {
      const int dhoff = kk * 32 + lh * 16;
      v16bf bk0 = *(const v16bf*)(kbase + (size_t)(kb + lr) * NQ + dhoff);
      v16bf bk1 = *(const v16bf*)(kbase + (size_t)(kb + 16 + lr) * NQ + dhoff);
      const v16bf a = (kk == 0) ? qa0.v : qa1.v;
      s0 = wmma_bf16(a, bk0, s0);
      s1 = wmma_bf16(a, bk1, s1);
    }
    const int key0 = kb + lr, key1 = kb + 16 + lr;
    float mx[8];
#pragma unroll
    for (int r = 0; r < 8; ++r) {
      const int q = qb * 16 + lh * 8 + r;
      float v0 = s0[r] * scale; if (key0 > q) v0 = -1e30f;
      float v1 = s1[r] * scale; if (key1 > q) v1 = -1e30f;
      s0[r] = v0; s1[r] = v1;
      mx[r] = fmaxf(v0, v1);
    }
#pragma unroll
    for (int d = 1; d < 16; d <<= 1)
#pragma unroll
      for (int r = 0; r < 8; ++r)
        mx[r] = fmaxf(mx[r], __shfl_xor(mx[r], d, 32));
    float al[8], rs[8];
#pragma unroll
    for (int r = 0; r < 8; ++r) {
      const float mn = fmaxf(m[r], mx[r]);
      al[r] = __expf(m[r] - mn);
      m[r] = mn;
      const float p0 = __expf(s0[r] - mn);
      const float p1 = __expf(s1[r] - mn);
      s0[r] = p0; s1[r] = p1;
      rs[r] = p0 + p1;
    }
#pragma unroll
    for (int d = 1; d < 16; d <<= 1)
#pragma unroll
      for (int r = 0; r < 8; ++r)
        rs[r] += __shfl_xor(rs[r], d, 32);
#pragma unroll
    for (int r = 0; r < 8; ++r) {
      l[r] = l[r] * al[r] + rs[r];
      o0[r] *= al[r]; o1[r] *= al[r]; o2[r] *= al[r]; o3[r] *= al[r];
      const int rowm = lh * 8 + r;
      ps[rowm * 32 + lr]      = (bf16_t)s0[r];
      ps[rowm * 32 + 16 + lr] = (bf16_t)s1[r];
    }
    // wave-local LDS transpose C-layout -> A-layout: wait for own DS stores
    asm volatile("s_wait_dscnt 0" ::: "memory");
    ABFrag pa;
    {
      const int kh = lh * 16;
      pa.h[0] = *(const v8bf*)(ps + lr * 32 + kh);
      pa.h[1] = *(const v8bf*)(ps + lr * 32 + kh + 8);
    }
#pragma unroll
    for (int t = 0; t < 4; ++t) {
      // B = V[32 keys x 64 dh]; lane holds dh column t*16+lr, K rows lh*16..+15
      const bf16_t* vcol = vbase + (size_t)(kb + lh * 16) * NQ + t * 16 + lr;
      v16bf vb;
#pragma unroll
      for (int e = 0; e < 16; ++e) vb[e] = vcol[(size_t)e * NQ];
      v8f& oo = (t == 0) ? o0 : (t == 1) ? o1 : (t == 2) ? o2 : o3;
      oo = wmma_bf16(pa.v, vb, oo);
    }
  }
#pragma unroll
  for (int r = 0; r < 8; ++r) {
    const float inv = 1.f / l[r];
    const int rowm = lh * 8 + r;
    const size_t rowoff = (size_t)(b * S + qb * 16 + rowm) * DM + h * DH;
    out[rowoff + 0  + lr] = (bf16_t)(o0[r] * inv);
    out[rowoff + 16 + lr] = (bf16_t)(o1[r] * inv);
    out[rowoff + 32 + lr] = (bf16_t)(o2[r] * inv);
    out[rowoff + 48 + lr] = (bf16_t)(o3[r] * inv);
  }
}

// ---------------------------------------------------------------------------
// Fused residual + LayerNorm: xo = xin + LN(y)*gamma + beta ; optional bf16 copy
// One block per row of D=1024.
// ---------------------------------------------------------------------------
__global__ __launch_bounds__(256) void ln_residual_kernel(
    const float* __restrict__ xin, const float* __restrict__ y,
    const float* __restrict__ gamma, const float* __restrict__ beta,
    float* __restrict__ xo, bf16_t* __restrict__ xob, int D) {
  const int row = blockIdx.x;
  const int tid = threadIdx.x;
  const float* yr = y + (size_t)row * D;
  __shared__ float red[256];
  float s = 0.f;
  for (int i = tid; i < D; i += 256) s += yr[i];
  red[tid] = s;
  __syncthreads();
  for (int st = 128; st > 0; st >>= 1) {
    if (tid < st) red[tid] += red[tid + st];
    __syncthreads();
  }
  const float mu = red[0] / (float)D;
  __syncthreads();
  float vs = 0.f;
  for (int i = tid; i < D; i += 256) {
    float d = yr[i] - mu;
    vs += d * d;
  }
  red[tid] = vs;
  __syncthreads();
  for (int st = 128; st > 0; st >>= 1) {
    if (tid < st) red[tid] += red[tid + st];
    __syncthreads();
  }
  const float rstd = rsqrtf(red[0] / (float)D + 1e-5f);
  const float* xr = xin + (size_t)row * D;
  for (int i = tid; i < D; i += 256) {
    float o = xr[i] + ((yr[i] - mu) * rstd * gamma[i] + beta[i]);
    xo[(size_t)row * D + i] = o;
    if (xob) xob[(size_t)row * D + i] = (bf16_t)o;
  }
}

// ---------------------------------------------------------------------------
extern "C" void kernel_launch(void* const* d_in, const int* in_sizes, int n_in,
                              void* d_out, int out_size, void* d_ws, size_t ws_size,
                              hipStream_t stream) {
  (void)in_sizes; (void)n_in; (void)out_size; (void)ws_size;
  constexpr int B = 2, S = 2048, D = 1024, H = 16, DH = 64, DMLP = 4096;
  constexpr int M = B * S;          // 4096 token rows
  constexpr int NQKV = 3 * H * DH;  // 3072

  const float* x     = (const float*)d_in[0];
  const float* wq    = (const float*)d_in[1];
  const float* wk    = (const float*)d_in[2];
  const float* wv    = (const float*)d_in[3];
  const float* wo    = (const float*)d_in[4];
  const float* w_in  = (const float*)d_in[5];
  const float* b_in  = (const float*)d_in[6];
  const float* w_out = (const float*)d_in[7];
  const float* b_out = (const float*)d_in[8];
  const float* g1    = (const float*)d_in[9];
  const float* bt1   = (const float*)d_in[10];
  const float* g2    = (const float*)d_in[11];
  const float* bt2   = (const float*)d_in[12];
  float* out = (float*)d_out;

  char* ws = (char*)d_ws;
  size_t off = 0;
  auto alloc = [&](size_t bytes) -> void* {
    void* p = ws + off;
    off = (off + bytes + 255) & ~(size_t)255;
    return p;
  };
  bf16_t* xb     = (bf16_t*)alloc((size_t)M * D * 2);
  bf16_t* wqkvT  = (bf16_t*)alloc((size_t)NQKV * D * 2);
  bf16_t* qkv    = (bf16_t*)alloc((size_t)M * NQKV * 2);
  bf16_t* attn   = (bf16_t*)alloc((size_t)M * D * 2);
  bf16_t* woT    = (bf16_t*)alloc((size_t)D * D * 2);
  bf16_t* winT   = (bf16_t*)alloc((size_t)DMLP * D * 2);
  bf16_t* woutT  = (bf16_t*)alloc((size_t)D * DMLP * 2);
  bf16_t* hid    = (bf16_t*)alloc((size_t)M * DMLP * 2);
  float*  proj   = (float*)alloc((size_t)M * D * 4);  // reused as mlp output
  float*  x2     = (float*)alloc((size_t)M * D * 4);
  bf16_t* x2b    = (bf16_t*)alloc((size_t)M * D * 2);

  // 1) pack inputs/weights to bf16 (B^T layouts for GEMM)
  {
    int n = M * D;
    cast_bf16_kernel<<<(n + 255) / 256, 256, 0, stream>>>(x, xb, n);
  }
  {
    int n = H * D * DH;  // 1,048,576
    pack_headw_kernel<<<(n + 255) / 256, 256, 0, stream>>>(wq, wqkvT + (size_t)0 * H * DH * D, D, DH, n);
    pack_headw_kernel<<<(n + 255) / 256, 256, 0, stream>>>(wk, wqkvT + (size_t)1 * H * DH * D, D, DH, n);
    pack_headw_kernel<<<(n + 255) / 256, 256, 0, stream>>>(wv, wqkvT + (size_t)2 * H * DH * D, D, DH, n);
  }
  {
    int n = D * D;
    transpose_cast_kernel<<<(n + 255) / 256, 256, 0, stream>>>(wo, woT, D, D);
  }
  {
    int n = D * DMLP;
    transpose_cast_kernel<<<(n + 255) / 256, 256, 0, stream>>>(w_in, winT, D, DMLP);
    transpose_cast_kernel<<<(n + 255) / 256, 256, 0, stream>>>(w_out, woutT, DMLP, D);
  }

  // 2) fused QKV projection: [M,1024] x [1024,3072] -> bf16 qkv
  gemm_bf16_kernel<<<dim3(NQKV / 64, M / 128), 256, 0, stream>>>(
      xb, wqkvT, nullptr, qkv, nullptr, M, NQKV, D, 0);

  // 3) causal flash attention (one wave per 16-query block)
  {
    int total_waves = B * H * (S / 16);  // 4096
    attn_kernel<<<total_waves / 8, 256, 0, stream>>>(qkv, attn);
  }

  // 4) output projection: [M,1024] x [1024,1024] -> f32 proj
  gemm_bf16_kernel<<<dim3(D / 64, M / 128), 256, 0, stream>>>(
      attn, woT, proj, nullptr, nullptr, M, D, D, 0);

  // 5) x2 = x + LN(proj); also bf16 copy for MLP
  ln_residual_kernel<<<M, 256, 0, stream>>>(x, proj, g1, bt1, x2, x2b, D);

  // 6) MLP up: relu(x2 @ w_in + b_in) -> bf16 hid
  gemm_bf16_kernel<<<dim3(DMLP / 64, M / 128), 256, 0, stream>>>(
      x2b, winT, nullptr, hid, b_in, M, DMLP, D, 1);

  // 7) MLP down: hid @ w_out + b_out -> f32 (reuse proj buffer)
  gemm_bf16_kernel<<<dim3(D / 64, M / 128), 256, 0, stream>>>(
      hid, woutT, proj, nullptr, b_out, M, D, DMLP, 0);

  // 8) out = x2 + LN(mlp)
  ln_residual_kernel<<<M, 256, 0, stream>>>(x2, proj, g2, bt2, out, nullptr, D);
}
